// PrunedInt4Linear_32779190403295
// MI455X (gfx1250) — matrix operand order repaired and checked by reading, not compiled, not run on hardware
//
#include <hip/hip_runtime.h>

// ---------------------------------------------------------------------------
// Fused int4-group-dequant linear:  y = x @ dequant(qweight)^T + bias
//   x: [8192, 4096] f32    qweight: [11008, 4096] int8 (codes -8..7)
//   scales: [11008, 64] f16    bias: [11008] f16    y: [8192, 11008] f32
//
// Compute-bound -> v_wmma_f32_16x16x32_bf16, 128x256 block tile, 64x64/wave,
// double-buffered LDS, one barrier per K-step. Raw weights staged with
// GLOBAL_LOAD_ASYNC_TO_LDS_B128 (ASYNCcnt) to free VGPRs and overlap copies.
// ---------------------------------------------------------------------------

#define M_TOTAL 8192
#define N_TOTAL 11008
#define K_TOTAL 4096
#define GROUP   64
#define NGROUPS (K_TOTAL / GROUP)

#define BM 128
#define BN 256
#define BK 64                 // == GROUP: one scale per (row, K-tile)
#define LSTR (BK + 8)         // bf16 tile row padding (16B) -> conflict-free
#define QSTR 80               // raw int8 row padding (80B) -> conflict-free
#define NK (K_TOTAL / BK)

typedef float  v8f    __attribute__((ext_vector_type(8)));
typedef __bf16 v16bf  __attribute__((ext_vector_type(16)));
typedef __bf16 bf16x8 __attribute__((ext_vector_type(8)));

__global__ __launch_bounds__(256)
void int4_linear_wmma_bf16(const float* __restrict__ x,
                           const signed char* __restrict__ qw,
                           const _Float16* __restrict__ scales,
                           const _Float16* __restrict__ bias,
                           float* __restrict__ y)
{
    // double-buffered bf16 tiles + raw int8 staging: 151,552 B total
    // -> 2 workgroups resident per 320KB WGP
    __shared__ __bf16      lds_x[2][BM][LSTR];
    __shared__ __bf16      lds_w[2][BN][LSTR];
    __shared__ signed char lds_q[2][BN][QSTR];

    const int t    = threadIdx.x;
    const int lane = t & 31;
    const int wave = t >> 5;
    const int waveM = wave >> 2;   // 0..1 -> 64 M-rows per wave
    const int waveN = wave & 3;    // 0..3 -> 64 N-cols per wave

    const long m_blk = (long)blockIdx.y * BM;
    const long n_blk = (long)blockIdx.x * BN;

    // 4x4 grid of 16x16 f32 accumulators (64x64 per wave)
    v8f acc[4][4] = {};

    // ---- staging coordinates (whole block cooperates) --------------------
    const int  xr = t >> 3;                                  // + p*32
    const int  xc = (t & 7) * 8;                             // 8 floats
    const long xbase = (m_blk + xr) * (long)K_TOTAL + xc;
    const long wbase = (n_blk + t) * (long)K_TOTAL;          // 1 weight row/thread

    bf16x8 xh[4];   // held x fragment (already bf16)
    float  sc;      // held group scale

    // issue global->LDS async copies for tile kt into staging buffer `buf`,
    // plus reg-staged x loads (x needs f32->bf16, so it cannot go async)
    auto load_tile = [&](int kt, int buf) {
        const long k0 = (long)kt * BK;
        // raw int8 weight row: 64B via 4 async b128 copies (VGPR-free path).
        // INST_OFFSET is added to both the LDS and global addresses.
        {
            const unsigned lq =
                (unsigned)(size_t)&lds_q[buf][t][0];       // DS address (low 32b)
            const unsigned long long ga =
                (unsigned long long)(qw + wbase + k0);
            asm volatile("global_load_async_to_lds_b128 %0, %1, off"
                         :: "v"(lq), "v"(ga) : "memory");
            asm volatile("global_load_async_to_lds_b128 %0, %1, off offset:16"
                         :: "v"(lq), "v"(ga) : "memory");
            asm volatile("global_load_async_to_lds_b128 %0, %1, off offset:32"
                         :: "v"(lq), "v"(ga) : "memory");
            asm volatile("global_load_async_to_lds_b128 %0, %1, off offset:48"
                         :: "v"(lq), "v"(ga) : "memory");
        }
        #pragma unroll
        for (int p = 0; p < 4; ++p) {
            const float4* src = reinterpret_cast<const float4*>(
                x + xbase + (long)p * 32 * K_TOTAL + k0);
            const float4 f0 = src[0];
            const float4 f1 = src[1];
            bf16x8 v;
            v[0] = (__bf16)f0.x; v[1] = (__bf16)f0.y;
            v[2] = (__bf16)f0.z; v[3] = (__bf16)f0.w;
            v[4] = (__bf16)f1.x; v[5] = (__bf16)f1.y;
            v[6] = (__bf16)f1.z; v[7] = (__bf16)f1.w;
            xh[p] = v;
        }
        sc = (float)scales[(n_blk + t) * NGROUPS + kt];
    };

    // write x tile from regs; dequantize this thread's own async-staged row
    auto store_tile = [&](int buf) {
        #pragma unroll
        for (int p = 0; p < 4; ++p)
            *reinterpret_cast<bf16x8*>(&lds_x[buf][p * 32 + xr][xc]) = xh[p];

        // wait for *our own* async copies; we only read the row we copied,
        // so no inter-wave barrier is needed here.
        asm volatile("s_wait_asynccnt 0" ::: "memory");

        union { uint4 v[4]; signed char b[64]; } q;
        const uint4* qr = reinterpret_cast<const uint4*>(&lds_q[buf][t][0]);
        #pragma unroll
        for (int p = 0; p < 4; ++p) q.v[p] = qr[p];
        #pragma unroll
        for (int c = 0; c < 8; ++c) {
            bf16x8 v;
            #pragma unroll
            for (int j = 0; j < 8; ++j)
                v[j] = (__bf16)((float)q.b[c * 8 + j] * sc);
            *reinterpret_cast<bf16x8*>(&lds_w[buf][t][c * 8]) = v;
        }
    };

    auto compute = [&](int buf) {
        #pragma unroll
        for (int kk = 0; kk < BK; kk += 32) {
            // A fragments: lane l -> M row (l%16), K chunks [base,base+8) and
            // [base+16,base+24), base = (l/16)*8   (ISA 16-bit A 16x32 layout)
            v16bf a[4];
            #pragma unroll
            for (int sm = 0; sm < 4; ++sm) {
                const int row = waveM * 64 + sm * 16 + (lane & 15);
                const int kb  = kk + ((lane >> 4) << 3);
                bf16x8 lo = *reinterpret_cast<const bf16x8*>(&lds_x[buf][row][kb]);
                bf16x8 hi = *reinterpret_cast<const bf16x8*>(&lds_x[buf][row][kb + 16]);
                a[sm] = __builtin_shufflevector(lo, hi,
                    0, 1, 2, 3, 4, 5, 6, 7, 8, 9, 10, 11, 12, 13, 14, 15);
            }
            // B fragments streamed one at a time to bound register pressure
            #pragma unroll
            for (int sn = 0; sn < 4; ++sn) {
                const int nrow = waveN * 64 + sn * 16 + (lane & 15);
                const int kb   = kk + ((lane >> 4) << 3);
                bf16x8 lo = *reinterpret_cast<const bf16x8*>(&lds_w[buf][nrow][kb]);
                bf16x8 hi = *reinterpret_cast<const bf16x8*>(&lds_w[buf][nrow][kb + 16]);
                v16bf b = __builtin_shufflevector(lo, hi,
                    0, 1, 2, 3, 4, 5, 6, 7, 8, 9, 10, 11, 12, 13, 14, 15);
                #pragma unroll
                for (int sm = 0; sm < 4; ++sm)
                    acc[sm][sn] = __builtin_amdgcn_wmma_f32_16x16x32_bf16(
                        false, a[sm], false, b,
                        (short)0, acc[sm][sn], false, false);
            }
        }
    };

    // ---- software-pipelined main loop: one barrier per K-step ------------
    load_tile(0, 0);
    store_tile(0);
    __syncthreads();

    #pragma unroll 1
    for (int kt = 0; kt < NK; ++kt) {
        const int cur = kt & 1;
        if (kt + 1 < NK) load_tile(kt + 1, cur ^ 1); // async + reg prefetch
        compute(cur);                                 // 32 x v_wmma per wave
        if (kt + 1 < NK) store_tile(cur ^ 1);         // dequant -> other buffer
        __syncthreads();
    }

    // ---- epilogue: add bias, store f32 ----------------------------------
    // C/D layout: VGPR r of lane l -> M = r + (l/16)*8, N = l%16
    #pragma unroll
    for (int sn = 0; sn < 4; ++sn) {
        const long n   = n_blk + waveN * 64 + sn * 16 + (lane & 15);
        const float bv = (float)bias[n];
        #pragma unroll
        for (int sm = 0; sm < 4; ++sm) {
            const long mbase = m_blk + waveM * 64 + sm * 16 + ((lane >> 4) << 3);
            #pragma unroll
            for (int r = 0; r < 8; ++r)
                y[(mbase + r) * (long)N_TOTAL + n] = acc[sm][sn][r] + bv;
        }
    }
}

extern "C" void kernel_launch(void* const* d_in, const int* in_sizes, int n_in,
                              void* d_out, int out_size, void* d_ws, size_t ws_size,
                              hipStream_t stream) {
    const float*       xp = (const float*)d_in[0];
    const signed char* qp = (const signed char*)d_in[1];
    const _Float16*    sp = (const _Float16*)d_in[2];
    const _Float16*    bp = (const _Float16*)d_in[3];
    float*             yp = (float*)d_out;

    dim3 grid(N_TOTAL / BN, M_TOTAL / BM);   // 43 x 64
    dim3 block(256);
    int4_linear_wmma_bf16<<<grid, block, 0, stream>>>(xp, qp, sp, bp, yp);
}
